// G2PGRU_82643760710287
// MI455X (gfx1250) — matrix operand order repaired and checked by reading, not compiled, not run on hardware
//
#include <hip/hip_runtime.h>
#include <hip/hip_bf16.h>
#include <cstdint>
#include <math.h>

// ---------------- problem constants ----------------
constexpr int NB  = 2048;   // batch
constexpr int NS  = 24;     // src seq len
constexpr int NT  = 25;     // trg seq len
constexpr int NH  = 256;    // hidden
constexpr int NE  = 300;    // embed
constexpr int NVO = 128;    // out vocab
constexpr int EP  = 320;    // E padded (mult of 64)
constexpr int G3  = 768;    // 3H
constexpr int H2  = 512;    // 2H
constexpr int XGP = 832;    // (E + 2H)=812 padded
constexpr int FCP = 1088;   // (H + 2H + E)=1068 padded

typedef __attribute__((ext_vector_type(16))) _Float16 v16h;
typedef __attribute__((ext_vector_type(8)))  _Float16 v8h;
typedef __attribute__((ext_vector_type(8)))  float    v8f;
typedef __attribute__((ext_vector_type(4)))  unsigned int u32x4;
typedef __attribute__((ext_vector_type(8)))  int i32x8;
typedef __attribute__((ext_vector_type(4)))  int i32x4;

// ---------------- WMMA GEMM with TDM double-buffered B staging ----------------
// C[M x N] = A[M x Kpad] (f16 row-major) * Bpacked (fragment layout) + bias
// grid.x = M/128 (8 waves per block, one 16-row mTile per wave)
// grid.y = N/64  (each wave covers 64 columns = 4 WMMA accumulators)
// B slab per kT (4 nTiles x 32 K = 4KB) is DMA'd global->LDS by the Tensor
// Data Mover (tensor_load_to_lds, TENSORcnt), double buffered.
__global__ __launch_bounds__(256)
void wmma_gemm_kernel(const _Float16* __restrict__ A,
                      const _Float16* __restrict__ Bp,
                      const float* __restrict__ bias,
                      float* __restrict__ C,
                      int Kpad, int N, int ldc)
{
    __shared__ _Float16 bsm[2][4 * 32 * 16];   // two 4KB slabs

    const int lane  = threadIdx.x & 31;
    const int wave  = threadIdx.x >> 5;
    const int mTile = blockIdx.x * 8 + wave;
    const int nBase = blockIdx.y * 64;
    const int numKT = Kpad >> 5;
    const int half  = lane >> 4;
    const int arow  = mTile * 16 + (lane & 15);
    const _Float16* Arow = A + (size_t)arow * Kpad;
    const int nT0 = nBase >> 4;

    // uniform TDM descriptor parts
    const uint64_t gbase   = (uint64_t)(uintptr_t)(Bp + (size_t)nT0 * numKT * 512);
    const uint32_t ldsbase = (uint32_t)(uintptr_t)(&bsm[0][0]); // flat LDS addr[31:0] = LDS offset

    auto tdm_issue = [&](int kt, int buf) {
        // 2D tile: rows = 4 nTiles (stride numKT*512 f16), row len = 512 f16 = 128 x 8B
        const uint64_t ga = gbase + (uint64_t)kt * 1024u;   // kt slab = 512 f16 = 1024B
        u32x4 g0;
        g0[0] = 1u;                                          // count=1, user mode
        g0[1] = ldsbase + (uint32_t)buf * 4096u;             // lds_addr
        g0[2] = (uint32_t)ga;                                // global_addr[31:0]
        g0[3] = (uint32_t)((ga >> 32) & 0x1FFFFFFu) | (2u << 30); // addr[56:32] | type=2
        i32x8 g1;
        g1[0] = (3 << 16);                                   // data_size = 8B units
        g1[1] = (128 << 16);                                 // tensor_dim0 = 128
        g1[2] = (4 << 16);                                   // tensor_dim1 = 4
        g1[3] = (128 << 16);                                 // tile_dim0 = 128
        g1[4] = 4;                                           // tile_dim1 = 4 (tile_dim2 = 0)
        g1[5] = numKT * 128;                                 // tensor_dim0_stride (8B units)
        g1[6] = 0;
        g1[7] = 0;
        i32x4 z4 = {0, 0, 0, 0};
        i32x8 z8 = {0, 0, 0, 0, 0, 0, 0, 0};
        __builtin_amdgcn_tensor_load_to_lds(g0, g1, z4, z4, z8, 0);
    };

    // prime slab 0
    if (wave == 0) {
        tdm_issue(0, 0);
        __builtin_amdgcn_s_wait_tensorcnt(0);
    }
    __syncthreads();

    v8f acc[4] = {};

    for (int kt = 0; kt < numKT; ++kt) {
        const int buf = kt & 1;
        if (wave == 0 && kt + 1 < numKT)
            tdm_issue(kt + 1, buf ^ 1);

        // ---- A fragment: 16x32 f16, lane=row, two adjacent 8-f16 chunks, k = 16*half + 0..15 ----
        const int k0 = kt * 32 + half * 16;
        v8h a0 = *(const v8h*)(Arow + k0);
        v8h a1 = *(const v8h*)(Arow + k0 + 8);
        v16h a;
        #pragma unroll
        for (int i = 0; i < 8; ++i) { a[i] = a0[i]; a[8 + i] = a1[i]; }

        #pragma unroll
        for (int j = 0; j < 4; ++j) {
            // ---- B fragment from LDS slab: [j][lane][16] -> 32 contiguous bytes ----
            const _Float16* bptr = &bsm[buf][(j * 32 + lane) * 16];
            v8h b0 = *(const v8h*)(bptr);
            v8h b1 = *(const v8h*)(bptr + 8);
            v16h b;
            #pragma unroll
            for (int i = 0; i < 8; ++i) { b[i] = b0[i]; b[8 + i] = b1[i]; }

            acc[j] = __builtin_amdgcn_wmma_f32_16x16x32_f16(
                false, a, false, b, (short)0, acc[j], false, false);
        }

        if (wave == 0 && kt + 1 < numKT)
            __builtin_amdgcn_s_wait_tensorcnt(0);
        __syncthreads();
    }

    // ---- C/D layout: VGPR r -> row r + 8*half, lane%16 -> col (ISA 7.12.2) ----
    const int rbase = mTile * 16 + half * 8;
    const int col0  = nBase + (lane & 15);
    #pragma unroll
    for (int j = 0; j < 4; ++j) {
        const int n  = col0 + j * 16;
        const float bv = bias ? bias[n] : 0.0f;
        #pragma unroll
        for (int r = 0; r < 8; ++r)
            C[(size_t)(rbase + r) * ldc + n] = acc[j][r] + bv;
    }
}

// ---------------- weight repack: W[K x N] f32 -> fragment-native f16 ----------------
__global__ void repack_w_kernel(const float* __restrict__ W, _Float16* __restrict__ out,
                                int K, int N, int Kpad)
{
    const int tid = blockIdx.x * blockDim.x + threadIdx.x;
    if (tid >= N * Kpad) return;
    const int j    = tid & 15;
    const int lane = (tid >> 4) & 31;
    const int rem  = tid >> 9;
    const int numKT = Kpad >> 5;
    const int kT = rem % numKT;
    const int nT = rem / numKT;
    const int n  = nT * 16 + (lane & 15);
    const int k  = kT * 32 + (lane >> 4) * 16 + j;
    float v = (k < K) ? W[(size_t)k * N + n] : 0.0f;
    out[tid] = (_Float16)v;
}

// ---------------- embeddings ----------------
__global__ void gather_enc_kernel(const int* __restrict__ src,
                                  const float* __restrict__ emb,
                                  _Float16* __restrict__ x16)
{
    const int idx = blockIdx.x * blockDim.x + threadIdx.x;
    if (idx >= NS * NB * EP) return;
    const int e = idx % EP;
    const int r = idx / EP;
    const int b = r % NB;
    const int s = r / NB;
    float v = 0.0f;
    if (e < NE) v = emb[(size_t)src[s * NB + b] * NE + e];
    x16[idx] = (_Float16)v;
}

__global__ void gather_dec_kernel(const int* __restrict__ trg_t,
                                  const float* __restrict__ emb,
                                  _Float16* __restrict__ e16)
{
    const int idx = blockIdx.x * blockDim.x + threadIdx.x;
    if (idx >= NB * NE) return;
    const int e = idx % NE;
    const int b = idx / NE;
    e16[idx] = (_Float16)emb[(size_t)trg_t[b] * NE + e];
}

// ---------------- GRU gate combine ----------------
__device__ __forceinline__ float sigmoidf(float x) { return 1.0f / (1.0f + expf(-x)); }

__global__ void gru_combine_kernel(const float* __restrict__ gi,
                                   const float* __restrict__ gh,
                                   float* __restrict__ h,
                                   _Float16* __restrict__ h16,
                                   _Float16* __restrict__ outseq) // pre-offset; may be null
{
    const int idx = blockIdx.x * blockDim.x + threadIdx.x;
    if (idx >= NB * NH) return;
    const int b = idx >> 8;
    const int i = idx & 255;
    const size_t base = (size_t)b * G3;
    const float ir = gi[base + i], iz = gi[base + NH + i], in_ = gi[base + 2 * NH + i];
    const float hr = gh[base + i], hz = gh[base + NH + i], hn  = gh[base + 2 * NH + i];
    const float r = sigmoidf(ir + hr);
    const float z = sigmoidf(iz + hz);
    const float n = tanhf(in_ + r * hn);
    const float hv = (1.0f - z) * n + z * h[idx];
    h[idx] = hv;
    h16[idx] = (_Float16)hv;
    if (outseq) outseq[(size_t)b * (NS * H2) + i] = (_Float16)hv;
}

// ---------------- misc elementwise ----------------
__global__ void zero_f32_kernel(float* p, int n)
{ int i = blockIdx.x * blockDim.x + threadIdx.x; if (i < n) p[i] = 0.0f; }

__global__ void zero_f16_kernel(_Float16* p, int n)
{ int i = blockIdx.x * blockDim.x + threadIdx.x; if (i < n) p[i] = (_Float16)0.0f; }

__global__ void concat_hfb_kernel(const _Float16* __restrict__ hA,
                                  const _Float16* __restrict__ hB,
                                  _Float16* __restrict__ cat)
{
    const int idx = blockIdx.x * blockDim.x + threadIdx.x;
    if (idx >= NB * H2) return;
    const int d = idx & 511;
    const int b = idx >> 9;
    cat[idx] = (d < NH) ? hA[b * NH + d] : hB[b * NH + d - NH];
}

__global__ void tanh_store_kernel(const float* __restrict__ t,
                                  float* __restrict__ h, _Float16* __restrict__ h16)
{
    const int idx = blockIdx.x * blockDim.x + threadIdx.x;
    if (idx >= NB * NH) return;
    const float v = tanhf(t[idx]);
    h[idx] = v;
    h16[idx] = (_Float16)v;
}

// ---------------- attention ----------------
__global__ __launch_bounds__(256)
void attn_scores_kernel(const float* __restrict__ hWh, const float* __restrict__ proj,
                        const float* __restrict__ v, float* __restrict__ scores)
{
    const int lane = threadIdx.x & 31;
    const int wid  = blockIdx.x * (blockDim.x >> 5) + (threadIdx.x >> 5);
    if (wid >= NB * NS) return;
    const int b = wid / NS, s = wid % NS;
    const float* hp = hWh + (size_t)b * NH;
    const float* pp = proj + ((size_t)b * NS + s) * NH;
    float sum = 0.0f;
    for (int i = lane; i < NH; i += 32) sum += tanhf(hp[i] + pp[i]) * v[i];
    #pragma unroll
    for (int o = 16; o >= 1; o >>= 1) sum += __shfl_xor(sum, o, 32);
    if (lane == 0) scores[wid] = sum;
}

__global__ __launch_bounds__(256)
void attn_softmax_kernel(const float* __restrict__ scores, float* __restrict__ aw)
{
    const int lane = threadIdx.x & 31;
    const int b    = blockIdx.x * (blockDim.x >> 5) + (threadIdx.x >> 5);
    if (b >= NB) return;
    float v = (lane < NS) ? scores[b * NS + lane] : -INFINITY;
    float m = v;
    #pragma unroll
    for (int o = 16; o >= 1; o >>= 1) m = fmaxf(m, __shfl_xor(m, o, 32));
    float e = (lane < NS) ? expf(v - m) : 0.0f;
    float sm = e;
    #pragma unroll
    for (int o = 16; o >= 1; o >>= 1) sm += __shfl_xor(sm, o, 32);
    if (lane < NS) aw[b * NS + lane] = e / sm;
}

__global__ void attn_weighted_kernel(const float* __restrict__ aw,
                                     const _Float16* __restrict__ enc_out,
                                     _Float16* __restrict__ w16)
{
    const int idx = blockIdx.x * blockDim.x + threadIdx.x;
    if (idx >= NB * H2) return;
    const int b = idx >> 9;
    const int d = idx & 511;
    const float* a = aw + (size_t)b * NS;
    const _Float16* eb = enc_out + (size_t)b * NS * H2 + d;
    float acc = 0.0f;
    #pragma unroll 4
    for (int s = 0; s < NS; ++s) acc += a[s] * (float)eb[(size_t)s * H2];
    w16[idx] = (_Float16)acc;
}

// ---------------- decoder operand assembly ----------------
__global__ void build_xg_kernel(const _Float16* __restrict__ emb,
                                const _Float16* __restrict__ wgt,
                                _Float16* __restrict__ xg)
{
    const int idx = blockIdx.x * blockDim.x + threadIdx.x;
    if (idx >= NB * XGP) return;
    const int k = idx % XGP;
    const int b = idx / XGP;
    _Float16 v = (_Float16)0.0f;
    if (k < NE)            v = emb[b * NE + k];
    else if (k < NE + H2)  v = wgt[b * H2 + (k - NE)];
    xg[idx] = v;
}

__global__ void build_fcA_kernel(const _Float16* __restrict__ h16,
                                 const _Float16* __restrict__ wgt,
                                 const _Float16* __restrict__ emb,
                                 _Float16* __restrict__ fa)
{
    const int idx = blockIdx.x * blockDim.x + threadIdx.x;
    if (idx >= NB * FCP) return;
    const int k = idx % FCP;
    const int b = idx / FCP;
    _Float16 v = (_Float16)0.0f;
    if (k < NH)                 v = h16[b * NH + k];
    else if (k < NH + H2)       v = wgt[b * H2 + (k - NH)];
    else if (k < NH + H2 + NE)  v = emb[b * NE + (k - NH - H2)];
    fa[idx] = v;
}

// ---------------- host-side launch helpers ----------------
static inline void gemm(hipStream_t st, const _Float16* A, const _Float16* Bp,
                        const float* bias, float* C, int M, int Kpad, int N, int ldc)
{
    dim3 g(M / 128, N / 64), blk(256);
    wmma_gemm_kernel<<<g, blk, 0, st>>>(A, Bp, bias, C, Kpad, N, ldc);
}

static inline void repack(hipStream_t st, const float* W, _Float16* out,
                          int K, int N, int Kpad)
{
    int tot = N * Kpad;
    repack_w_kernel<<<(tot + 255) / 256, 256, 0, st>>>(W, out, K, N, Kpad);
}

extern "C" void kernel_launch(void* const* d_in, const int* in_sizes, int n_in,
                              void* d_out, int out_size, void* d_ws, size_t ws_size,
                              hipStream_t stream)
{
    // ------- inputs (setup_inputs dict order) -------
    const int*   src       = (const int*)  d_in[0];
    const int*   trg       = (const int*)  d_in[1];
    const float* enc_emb   = (const float*)d_in[2];
    const float* enc_Wih_f = (const float*)d_in[3];
    const float* enc_Whh_f = (const float*)d_in[4];
    const float* enc_bih_f = (const float*)d_in[5];
    const float* enc_bhh_f = (const float*)d_in[6];
    const float* enc_Wih_b = (const float*)d_in[7];
    const float* enc_Whh_b = (const float*)d_in[8];
    const float* enc_bih_b = (const float*)d_in[9];
    const float* enc_bhh_b = (const float*)d_in[10];
    const float* enc_fcW   = (const float*)d_in[11];
    const float* enc_fcb   = (const float*)d_in[12];
    const float* attn_Wh   = (const float*)d_in[13];
    const float* attn_We   = (const float*)d_in[14];
    const float* attn_b    = (const float*)d_in[15];
    const float* attn_v    = (const float*)d_in[16];
    const float* dec_emb   = (const float*)d_in[17];
    const float* dec_Wih   = (const float*)d_in[18];
    const float* dec_Whh   = (const float*)d_in[19];
    const float* dec_bih   = (const float*)d_in[20];
    const float* dec_bhh   = (const float*)d_in[21];
    const float* fcW       = (const float*)d_in[22];
    const float* fcb       = (const float*)d_in[23];
    float* out = (float*)d_out;

    // ------- workspace bump allocator -------
    char* w = (char*)d_ws;
    size_t off = 0;
    auto alloc = [&](size_t bytes) -> void* {
        void* p = w + off;
        off = (off + bytes + 255) & ~(size_t)255;
        return p;
    };
    _Float16* pIhF  = (_Float16*)alloc((size_t)EP  * G3  * 2);
    _Float16* pHhF  = (_Float16*)alloc((size_t)NH  * G3  * 2);
    _Float16* pIhB  = (_Float16*)alloc((size_t)EP  * G3  * 2);
    _Float16* pHhB  = (_Float16*)alloc((size_t)NH  * G3  * 2);
    _Float16* pFcE  = (_Float16*)alloc((size_t)H2  * NH  * 2);
    _Float16* pAWh  = (_Float16*)alloc((size_t)NH  * NH  * 2);
    _Float16* pAWe  = (_Float16*)alloc((size_t)H2  * NH  * 2);
    _Float16* pDIh  = (_Float16*)alloc((size_t)XGP * G3  * 2);
    _Float16* pDHh  = (_Float16*)alloc((size_t)NH  * G3  * 2);
    _Float16* pFc   = (_Float16*)alloc((size_t)FCP * NVO * 2);
    _Float16* x16     = (_Float16*)alloc((size_t)NS * NB * EP * 2);
    _Float16* encOut  = (_Float16*)alloc((size_t)NB * NS * H2 * 2);
    float*    projF   = (float*)   alloc((size_t)NB * NS * NH * 4);
    float*    gi      = (float*)   alloc((size_t)NB * G3 * 4);
    float*    gh      = (float*)   alloc((size_t)NB * G3 * 4);
    float*    hA      = (float*)   alloc((size_t)NB * NH * 4);
    _Float16* hA16    = (_Float16*)alloc((size_t)NB * NH * 2);
    float*    hB      = (float*)   alloc((size_t)NB * NH * 4);
    _Float16* hB16    = (_Float16*)alloc((size_t)NB * NH * 2);
    float*    hD      = (float*)   alloc((size_t)NB * NH * 4);
    _Float16* hD16    = (_Float16*)alloc((size_t)NB * NH * 2);
    _Float16* cat16   = (_Float16*)alloc((size_t)NB * H2 * 2);
    float*    hWh     = (float*)   alloc((size_t)NB * NH * 4);
    float*    scores  = (float*)   alloc((size_t)NB * NS * 4);
    float*    attnw   = (float*)   alloc((size_t)NB * NS * 4);
    _Float16* wgt16   = (_Float16*)alloc((size_t)NB * H2 * 2);
    _Float16* emb16   = (_Float16*)alloc((size_t)NB * NE * 2);
    _Float16* xg16    = (_Float16*)alloc((size_t)NB * XGP * 2);
    _Float16* fcA16   = (_Float16*)alloc((size_t)NB * FCP * 2);
    (void)ws_size; (void)n_in; (void)in_sizes; (void)out_size;

    // ------- weight repack (once per launch; tiny) -------
    repack(stream, enc_Wih_f, pIhF, NE,           G3,  EP);
    repack(stream, enc_Whh_f, pHhF, NH,           G3,  NH);
    repack(stream, enc_Wih_b, pIhB, NE,           G3,  EP);
    repack(stream, enc_Whh_b, pHhB, NH,           G3,  NH);
    repack(stream, enc_fcW,   pFcE, H2,           NH,  H2);
    repack(stream, attn_Wh,   pAWh, NH,           NH,  NH);
    repack(stream, attn_We,   pAWe, H2,           NH,  H2);
    repack(stream, dec_Wih,   pDIh, NE + H2,      G3,  XGP);
    repack(stream, dec_Whh,   pDHh, NH,           G3,  NH);
    repack(stream, fcW,       pFc,  NH + H2 + NE, NVO, FCP);

    // ------- encoder embeddings + h0 + outputs[0]=0 -------
    {
        int tot = NS * NB * EP;
        gather_enc_kernel<<<(tot + 255) / 256, 256, 0, stream>>>(src, enc_emb, x16);
        zero_f32_kernel<<<(NB * NH + 255) / 256, 256, 0, stream>>>(hA, NB * NH);
        zero_f16_kernel<<<(NB * NH + 255) / 256, 256, 0, stream>>>(hA16, NB * NH);
        zero_f32_kernel<<<(NB * NH + 255) / 256, 256, 0, stream>>>(hB, NB * NH);
        zero_f16_kernel<<<(NB * NH + 255) / 256, 256, 0, stream>>>(hB16, NB * NH);
        zero_f32_kernel<<<(NB * NVO + 255) / 256, 256, 0, stream>>>(out, NB * NVO);
    }

    const int nBH = NB * NH;
    // ------- forward encoder scan -------
    for (int s = 0; s < NS; ++s) {
        gemm(stream, x16 + (size_t)s * NB * EP, pIhF, enc_bih_f, gi, NB, EP, G3, G3);
        gemm(stream, hA16,                      pHhF, enc_bhh_f, gh, NB, NH, G3, G3);
        gru_combine_kernel<<<(nBH + 255) / 256, 256, 0, stream>>>(
            gi, gh, hA, hA16, encOut + (size_t)s * H2);
    }
    // ------- backward encoder scan -------
    for (int i = 0; i < NS; ++i) {
        const int s = NS - 1 - i;
        gemm(stream, x16 + (size_t)s * NB * EP, pIhB, enc_bih_b, gi, NB, EP, G3, G3);
        gemm(stream, hB16,                      pHhB, enc_bhh_b, gh, NB, NH, G3, G3);
        gru_combine_kernel<<<(nBH + 255) / 256, 256, 0, stream>>>(
            gi, gh, hB, hB16, encOut + (size_t)s * H2 + NH);
    }
    // ------- hidden = tanh(concat(hf,hb) @ enc_fcW + enc_fcb) -------
    concat_hfb_kernel<<<(NB * H2 + 255) / 256, 256, 0, stream>>>(hA16, hB16, cat16);
    gemm(stream, cat16, pFcE, enc_fcb, gi /*tmp*/, NB, H2, NH, NH);
    tanh_store_kernel<<<(nBH + 255) / 256, 256, 0, stream>>>(gi, hD, hD16);
    // ------- enc_proj = enc_bse @ attn_We + attn_b (bias folded) -------
    gemm(stream, encOut, pAWe, attn_b, projF, NB * NS, H2, NH, NH);

    // ------- decoder scan -------
    for (int t = 0; t < NT - 1; ++t) {
        gather_dec_kernel<<<(NB * NE + 255) / 256, 256, 0, stream>>>(
            trg + (size_t)t * NB, dec_emb, emb16);
        gemm(stream, hD16, pAWh, nullptr, hWh, NB, NH, NH, NH);
        attn_scores_kernel<<<(NB * NS) / 8, 256, 0, stream>>>(hWh, projF, attn_v, scores);
        attn_softmax_kernel<<<NB / 8, 256, 0, stream>>>(scores, attnw);
        attn_weighted_kernel<<<(NB * H2 + 255) / 256, 256, 0, stream>>>(attnw, encOut, wgt16);
        build_xg_kernel<<<(NB * XGP + 255) / 256, 256, 0, stream>>>(emb16, wgt16, xg16);
        gemm(stream, xg16, pDIh, dec_bih, gi, NB, XGP, G3, G3);
        gemm(stream, hD16, pDHh, dec_bhh, gh, NB, NH, G3, G3);
        gru_combine_kernel<<<(nBH + 255) / 256, 256, 0, stream>>>(gi, gh, hD, hD16, nullptr);
        build_fcA_kernel<<<(NB * FCP + 255) / 256, 256, 0, stream>>>(hD16, wgt16, emb16, fcA16);
        gemm(stream, fcA16, pFc, fcb, out + (size_t)(t + 1) * NB * NVO, NB, FCP, NVO, NVO);
    }
}
